// GNN_73495480369262
// MI455X (gfx1250) — matrix operand order repaired and checked by reading, not compiled, not run on hardware
//
#include <hip/hip_runtime.h>
#include <hip/hip_bf16.h>

typedef __attribute__((ext_vector_type(16))) _Float16 v16h;
typedef __attribute__((ext_vector_type(8)))  float    v8f;

#define N_USERS  200000
#define N_MOVIES 80000
#define N_EDGE   2000000
#define N_LBL    500000
#define DD 128
#define HH 64

// ---------------------------------------------------------------------------
// WMMA wrapper: D = A(16x32 f16) * B(32x16 f16) + C(16x16 f32)
// ---------------------------------------------------------------------------
__device__ __forceinline__ v8f wmma16x16x32(v16h a, v16h b, v8f c) {
    return __builtin_amdgcn_wmma_f32_16x16x32_f16(
        /*neg_a=*/false, a, /*neg_b=*/false, b,
        /*c_mod=*/(short)0, c, /*reuse_a=*/false, /*reuse_b=*/false);
}

// ---------------------------------------------------------------------------
// A-fragment load: 16x32 tile from row-major f32 matrix, per CDNA5 16-bit A
// layout.  lane<16: m=lane, holds K = k0..k0+7 , k0+16..k0+23
//          lane>=16: m=lane-16, holds K = k0+8..k0+15, k0+24..k0+31
// ---------------------------------------------------------------------------
__device__ __forceinline__ v16h load_a_frag(const float* __restrict__ row,
                                            int k0, float scale, int lane) {
    const int base = k0 + ((lane & 16) ? 8 : 0);
    const float4 f0 = *(const float4*)(row + base);
    const float4 f1 = *(const float4*)(row + base + 4);
    const float4 f2 = *(const float4*)(row + base + 16);
    const float4 f3 = *(const float4*)(row + base + 20);
    v16h a;
    a[0]  = (_Float16)(f0.x * scale); a[1]  = (_Float16)(f0.y * scale);
    a[2]  = (_Float16)(f0.z * scale); a[3]  = (_Float16)(f0.w * scale);
    a[4]  = (_Float16)(f1.x * scale); a[5]  = (_Float16)(f1.y * scale);
    a[6]  = (_Float16)(f1.z * scale); a[7]  = (_Float16)(f1.w * scale);
    a[8]  = (_Float16)(f2.x * scale); a[9]  = (_Float16)(f2.y * scale);
    a[10] = (_Float16)(f2.z * scale); a[11] = (_Float16)(f2.w * scale);
    a[12] = (_Float16)(f3.x * scale); a[13] = (_Float16)(f3.y * scale);
    a[14] = (_Float16)(f3.z * scale); a[15] = (_Float16)(f3.w * scale);
    return a;
}

// B-fragment: pre-packed so each lane's 16 halfs are contiguous (32B load).
__device__ __forceinline__ v16h load_b_frag(const _Float16* __restrict__ P,
                                            int kt, int nt, int ntiles_n, int lane) {
    const _Float16* p = P + (((size_t)(kt * ntiles_n + nt) * 32 + lane) << 4);
    return *(const v16h*)p;
}

// ---------------------------------------------------------------------------
// Pack f32 weight [K,N] into B-fragment layout (f16).
// packed idx p = ((kt*NT + nt)*32 + l)*16 + i  ->  k = kt*32 + (l/16)*8 + (i%8) + (i/8)*16,
//                                                  n = nt*16 + (l%16)
// ---------------------------------------------------------------------------
__global__ void pack_w_kernel(const float* __restrict__ W, _Float16* __restrict__ P,
                              int K, int N) {
    int p = blockIdx.x * blockDim.x + threadIdx.x;
    if (p >= K * N) return;
    int i  = p & 15;
    int l  = (p >> 4) & 31;
    int nt = (p >> 9) % (N >> 4);
    int kt = p / ((N >> 4) << 9);
    int k  = kt * 32 + ((l >> 4) << 3) + (i & 7) + ((i >> 3) << 4);
    int n  = nt * 16 + (l & 15);
    P[p] = (_Float16)W[(size_t)k * N + n];
}

// ---------------------------------------------------------------------------
// Edge degree counts (float, for later reciprocal)
// ---------------------------------------------------------------------------
__global__ void count_kernel(const int* __restrict__ src, const int* __restrict__ dst,
                             float* __restrict__ cnt_u, float* __restrict__ cnt_m, int E) {
    int e = blockIdx.x * blockDim.x + threadIdx.x;
    if (e >= E) return;
    atomicAdd(&cnt_u[src[e]], 1.0f);
    atomicAdd(&cnt_m[dst[e]], 1.0f);
}

// Scatter both directions, F floats per row, 4 floats per thread.
template<int F>
__global__ void scatter_kernel(const float* __restrict__ x_src_side,  // [.., F] indexed by src
                               const float* __restrict__ x_dst_side,  // [.., F] indexed by dst
                               const int* __restrict__ src, const int* __restrict__ dst,
                               float* __restrict__ sum_at_dst,        // [.., F]
                               float* __restrict__ sum_at_src,        // [.., F]
                               int E) {
    const int chunks = F / 4;
    unsigned gid = blockIdx.x * blockDim.x + threadIdx.x;
    int e = gid / chunks;
    if (e >= E) return;
    int c = (gid % chunks) * 4;
    int s = src[e], d = dst[e];
    float4 vs = *(const float4*)(x_src_side + (size_t)s * F + c);
    float4 vd = *(const float4*)(x_dst_side + (size_t)d * F + c);
    float* pm = sum_at_dst + (size_t)d * F + c;
    float* pu = sum_at_src + (size_t)s * F + c;
    atomicAdd(pm + 0, vs.x); atomicAdd(pm + 1, vs.y);
    atomicAdd(pm + 2, vs.z); atomicAdd(pm + 3, vs.w);
    atomicAdd(pu + 0, vd.x); atomicAdd(pu + 1, vd.y);
    atomicAdd(pu + 2, vd.z); atomicAdd(pu + 3, vd.w);
}

// ---------------------------------------------------------------------------
// Fused SAGE node update:  Out = act( (sum/deg) @ Wl + b + Xdst @ Wr )
// One wave -> 16 rows x 64 cols.  K in {128, 64}.
// ---------------------------------------------------------------------------
template<int K, bool RELU>
__launch_bounds__(128)
__global__ void sage_gemm_kernel(const float* __restrict__ Xagg, const float* __restrict__ cnt,
                                 const _Float16* __restrict__ Wl, const float* __restrict__ bias,
                                 const float* __restrict__ Xdst, const _Float16* __restrict__ Wr,
                                 float* __restrict__ Out, int M) {
    const int lane = threadIdx.x & 31;
    const int wave = (blockIdx.x * (blockDim.x >> 5)) + (threadIdx.x >> 5);
    const int rowBase = wave * 16;
    if (rowBase >= M) return;

    const int m = lane & 15;
    const float deg = cnt[rowBase + m];
    const float rscale = 1.0f / fmaxf(deg, 1.0f);

    v8f acc[4] = {v8f{}, v8f{}, v8f{}, v8f{}};

    #pragma unroll
    for (int op = 0; op < 2; ++op) {
        const float*     X = op ? Xdst : Xagg;
        const _Float16*  W = op ? Wr   : Wl;
        const float  scale = op ? 1.0f : rscale;
        const float*   row = X + (size_t)(rowBase + m) * K;
        #pragma unroll
        for (int kt = 0; kt < K / 32; ++kt) {
            v16h a = load_a_frag(row, kt * 32, scale, lane);
            #pragma unroll
            for (int nt = 0; nt < 4; ++nt) {
                v16h b = load_b_frag(W, kt, nt, 4, lane);
                acc[nt] = wmma16x16x32(a, b, acc[nt]);
            }
        }
    }

    // C layout: vgpr r, lanes0-15 -> M=r, lanes16-31 -> M=8+r; N = nt*16 + lane%16
    const int n0 = lane & 15;
    const int rowOff = rowBase + ((lane >> 4) << 3);
    #pragma unroll
    for (int nt = 0; nt < 4; ++nt) {
        const int n = nt * 16 + n0;
        const float bn = bias[n];
        #pragma unroll
        for (int r = 0; r < 8; ++r) {
            float v = acc[nt][r] + bn;
            if (RELU) v = fmaxf(v, 0.0f);
            Out[(size_t)(rowOff + r) * HH + n] = v;
        }
    }
}

// ---------------------------------------------------------------------------
// Edge decoder: out = relu([z_u[ls] | z_m[ld]] @ Wd1 + bd1) @ Wd2 + bd2
// One wave -> 16 labels.  K-tiles 0..1 gather z_u, 2..3 gather z_m.
// ---------------------------------------------------------------------------
__launch_bounds__(128)
__global__ void decoder_kernel(const float* __restrict__ z_u, const float* __restrict__ z_m,
                               const int* __restrict__ lsrc, const int* __restrict__ ldst,
                               const _Float16* __restrict__ Wd1p, const float* __restrict__ bd1,
                               const float* __restrict__ Wd2, const float* __restrict__ bd2,
                               float* __restrict__ out, int L) {
    const int lane = threadIdx.x & 31;
    const int wave = (blockIdx.x * (blockDim.x >> 5)) + (threadIdx.x >> 5);
    const int rowBase = wave * 16;
    if (rowBase >= L) return;

    const int m = lane & 15;
    v8f acc[4] = {v8f{}, v8f{}, v8f{}, v8f{}};

    #pragma unroll
    for (int kt = 0; kt < 4; ++kt) {
        const int k0 = kt * 32;
        const float* row;
        if (k0 < HH) row = z_u + (size_t)lsrc[rowBase + m] * HH + k0;
        else         row = z_m + (size_t)ldst[rowBase + m] * HH + (k0 - HH);
        v16h a = load_a_frag(row, 0, 1.0f, lane);
        #pragma unroll
        for (int nt = 0; nt < 4; ++nt) {
            v16h b = load_b_frag(Wd1p, kt, nt, 4, lane);
            acc[nt] = wmma16x16x32(a, b, acc[nt]);
        }
    }

    // bias + relu + dot(Wd2) directly from C fragments
    const int n0 = lane & 15;
    float part[8];
    #pragma unroll
    for (int r = 0; r < 8; ++r) part[r] = 0.0f;
    #pragma unroll
    for (int nt = 0; nt < 4; ++nt) {
        const float bn = bd1[nt * 16 + n0];
        const float wn = Wd2[nt * 16 + n0];
        #pragma unroll
        for (int r = 0; r < 8; ++r)
            part[r] += fmaxf(acc[nt][r] + bn, 0.0f) * wn;
    }
    // reduce across the 16 lanes of each half-wave (bits 0..3 stay in-half)
    #pragma unroll
    for (int mask = 1; mask < 16; mask <<= 1) {
        #pragma unroll
        for (int r = 0; r < 8; ++r)
            part[r] += __shfl_xor(part[r], mask, 32);
    }
    if (n0 == 0) {
        const float b2 = bd2[0];
        const int mbase = rowBase + ((lane >> 4) << 3);
        #pragma unroll
        for (int r = 0; r < 8; ++r)
            out[mbase + r] = part[r] + b2;
    }
}

// ---------------------------------------------------------------------------
// Host launcher
// ---------------------------------------------------------------------------
extern "C" void kernel_launch(void* const* d_in, const int* in_sizes, int n_in,
                              void* d_out, int out_size, void* d_ws, size_t ws_size,
                              hipStream_t stream) {
    const float* x_user  = (const float*)d_in[0];
    const float* x_movie = (const float*)d_in[1];
    const int* edge_src  = (const int*)d_in[2];
    const int* edge_dst  = (const int*)d_in[3];
    const int* lbl_src   = (const int*)d_in[4];
    const int* lbl_dst   = (const int*)d_in[5];
    const float* W1_u2m_l = (const float*)d_in[6];
    const float* b1_u2m   = (const float*)d_in[7];
    const float* W1_u2m_r = (const float*)d_in[8];
    const float* W1_m2u_l = (const float*)d_in[9];
    const float* b1_m2u   = (const float*)d_in[10];
    const float* W1_m2u_r = (const float*)d_in[11];
    const float* W2_u2m_l = (const float*)d_in[12];
    const float* b2_u2m   = (const float*)d_in[13];
    const float* W2_u2m_r = (const float*)d_in[14];
    const float* W2_m2u_l = (const float*)d_in[15];
    const float* b2_m2u   = (const float*)d_in[16];
    const float* W2_m2u_r = (const float*)d_in[17];
    const float* Wd1      = (const float*)d_in[18];
    const float* bd1      = (const float*)d_in[19];
    const float* Wd2      = (const float*)d_in[20];
    const float* bd2      = (const float*)d_in[21];
    float* out = (float*)d_out;

    // ---- workspace layout (floats) ----
    float* ws    = (float*)d_ws;
    float* cnt_u = ws;                                   // N_USERS
    float* cnt_m = cnt_u + N_USERS;                      // N_MOVIES
    float* sum_u = cnt_m + N_MOVIES;                     // N_USERS*128 (reused for layer2 @64)
    float* sum_m = sum_u + (size_t)N_USERS * DD;         // N_MOVIES*128 (reused)
    float* h_u   = sum_m + (size_t)N_MOVIES * DD;        // N_USERS*64
    float* h_m   = h_u + (size_t)N_USERS * HH;           // N_MOVIES*64
    float* z_u   = h_m + (size_t)N_MOVIES * HH;          // N_USERS*64
    float* z_m   = z_u + (size_t)N_USERS * HH;           // N_MOVIES*64
    _Float16* p_W1ul = (_Float16*)(z_m + (size_t)N_MOVIES * HH);
    _Float16* p_W1ur = p_W1ul + DD * HH;
    _Float16* p_W1ml = p_W1ur + DD * HH;
    _Float16* p_W1mr = p_W1ml + DD * HH;
    _Float16* p_W2ul = p_W1mr + DD * HH;
    _Float16* p_W2ur = p_W2ul + HH * HH;
    _Float16* p_W2ml = p_W2ur + HH * HH;
    _Float16* p_W2mr = p_W2ml + HH * HH;
    _Float16* p_Wd1  = p_W2mr + HH * HH;

    auto cdiv = [](long long a, long long b) { return (int)((a + b - 1) / b); };

    // ---- pack weights into WMMA B-fragment layout ----
    pack_w_kernel<<<cdiv(DD * HH, 256), 256, 0, stream>>>(W1_u2m_l, p_W1ul, DD, HH);
    pack_w_kernel<<<cdiv(DD * HH, 256), 256, 0, stream>>>(W1_u2m_r, p_W1ur, DD, HH);
    pack_w_kernel<<<cdiv(DD * HH, 256), 256, 0, stream>>>(W1_m2u_l, p_W1ml, DD, HH);
    pack_w_kernel<<<cdiv(DD * HH, 256), 256, 0, stream>>>(W1_m2u_r, p_W1mr, DD, HH);
    pack_w_kernel<<<cdiv(HH * HH, 256), 256, 0, stream>>>(W2_u2m_l, p_W2ul, HH, HH);
    pack_w_kernel<<<cdiv(HH * HH, 256), 256, 0, stream>>>(W2_u2m_r, p_W2ur, HH, HH);
    pack_w_kernel<<<cdiv(HH * HH, 256), 256, 0, stream>>>(W2_m2u_l, p_W2ml, HH, HH);
    pack_w_kernel<<<cdiv(HH * HH, 256), 256, 0, stream>>>(W2_m2u_r, p_W2mr, HH, HH);
    pack_w_kernel<<<cdiv(DD * HH, 256), 256, 0, stream>>>(Wd1, p_Wd1, DD, HH);

    // ---- zero counts + layer-1 sums (contiguous region) ----
    size_t zero1 = (size_t)(N_USERS + N_MOVIES) * sizeof(float)
                 + ((size_t)N_USERS * DD + (size_t)N_MOVIES * DD) * sizeof(float);
    hipMemsetAsync(ws, 0, zero1, stream);

    // ---- degrees ----
    count_kernel<<<cdiv(N_EDGE, 256), 256, 0, stream>>>(edge_src, edge_dst, cnt_u, cnt_m, N_EDGE);

    // ---- layer 1: scatter x_user->sum_m (by dst), x_movie->sum_u (by src) ----
    scatter_kernel<DD><<<cdiv((long long)N_EDGE * (DD / 4), 256), 256, 0, stream>>>(
        x_user, x_movie, edge_src, edge_dst, sum_m, sum_u, N_EDGE);

    // ---- layer 1 node updates (relu) ----
    sage_gemm_kernel<DD, true><<<cdiv(N_MOVIES / 16, 4), 128, 0, stream>>>(
        sum_m, cnt_m, p_W1ul, b1_u2m, x_movie, p_W1ur, h_m, N_MOVIES);
    sage_gemm_kernel<DD, true><<<cdiv(N_USERS / 16, 4), 128, 0, stream>>>(
        sum_u, cnt_u, p_W1ml, b1_m2u, x_user, p_W1mr, h_u, N_USERS);

    // ---- layer 2: reuse sum buffers at width 64 ----
    hipMemsetAsync(sum_u, 0, (size_t)N_USERS * HH * sizeof(float), stream);
    hipMemsetAsync(sum_m, 0, (size_t)N_MOVIES * HH * sizeof(float), stream);
    scatter_kernel<HH><<<cdiv((long long)N_EDGE * (HH / 4), 256), 256, 0, stream>>>(
        h_u, h_m, edge_src, edge_dst, sum_m, sum_u, N_EDGE);

    sage_gemm_kernel<HH, false><<<cdiv(N_MOVIES / 16, 4), 128, 0, stream>>>(
        sum_m, cnt_m, p_W2ul, b2_u2m, h_m, p_W2ur, z_m, N_MOVIES);
    sage_gemm_kernel<HH, false><<<cdiv(N_USERS / 16, 4), 128, 0, stream>>>(
        sum_u, cnt_u, p_W2ml, b2_m2u, h_u, p_W2mr, z_u, N_USERS);

    // ---- decoder ----
    decoder_kernel<<<cdiv(N_LBL / 16, 4), 128, 0, stream>>>(
        z_u, z_m, lbl_src, lbl_dst, p_Wd1, bd1, Wd2, bd2, out, N_LBL);
}
